// GCN_32744830665494
// MI455X (gfx1250) — matrix operand order repaired and checked by reading, not compiled, not run on hardware
//
#include <hip/hip_runtime.h>

typedef __attribute__((ext_vector_type(16))) __bf16   v16bf;
typedef __attribute__((ext_vector_type(8)))  float    v8f;
typedef __attribute__((ext_vector_type(4)))  unsigned int u32x4;

// float -> bf16 (round to nearest even, NaN-safe), stored as u16
static __device__ __forceinline__ unsigned short f2bf(float f) {
    unsigned int u = __float_as_uint(f);
    if ((u & 0x7fffffffu) > 0x7f800000u) return (unsigned short)((u >> 16) | 0x0040u);
    unsigned int r = u + 0x7fffu + ((u >> 16) & 1u);
    return (unsigned short)(r >> 16);
}

__global__ void k_zero(float* __restrict__ p, long n) {
    long i = blockIdx.x * (long)blockDim.x + threadIdx.x;
    if (i < n) p[i] = 0.0f;
}

__global__ void k_degree(const int* __restrict__ dst, float* __restrict__ deg, int E) {
    int e = blockIdx.x * blockDim.x + threadIdx.x;
    if (e < E) atomicAdd(&deg[dst[e]], 1.0f);
}

__global__ void k_dinv(const float* __restrict__ deg, float* __restrict__ dinv, int n) {
    int i = blockIdx.x * blockDim.x + threadIdx.x;
    if (i < n) dinv[i] = rsqrtf(deg[i] + 1.0f);   // +1 = self-loop
}

__global__ void k_cvt_bf16(const float* __restrict__ s, unsigned short* __restrict__ d, long n) {
    long i = blockIdx.x * (long)blockDim.x + threadIdx.x;
    if (i < n) d[i] = f2bf(s[i]);
}

// W [fi x fo] fp32 row-major -> Wt [fo x fi] bf16 row-major (B-fragment friendly)
__global__ void k_cvt_w(const float* __restrict__ W, unsigned short* __restrict__ Wt,
                        int fi, int fo) {
    int t = blockIdx.x * blockDim.x + threadIdx.x;
    if (t < fi * fo) {
        int i = t / fo, o = t % fo;
        Wt[(long)o * fi + i] = f2bf(W[t]);
    }
}

// H[nrows x fo] = A[nrows x fi](bf16) @ Wt^T.
// One wave computes a 16x64 tile (4 adjacent 16x16 tiles sharing one A fragment),
// v_wmma_f32_16x16x32_bf16 over K in steps of 32.
__global__ void k_gemm_wmma(const unsigned short* __restrict__ A,
                            const unsigned short* __restrict__ Bt,
                            float* __restrict__ H, int nrows, int fi, int fo) {
    int wid  = (blockIdx.x * blockDim.x + threadIdx.x) >> 5;
    int lane = threadIdx.x & 31;
    int ntb  = fo >> 6;                       // # of 64-wide column blocks
    int tm   = wid / ntb;                     // 16-row tile index
    int tb   = wid - tm * ntb;                // 64-col block index
    if (tm * 16 >= nrows) return;             // wave-uniform: EXEC stays all-1s
    int lhi = lane >> 4, llo = lane & 15;

    // A fragment: lane row = llo; elems 0-15 -> K = k + lhi*16 + e
    const unsigned short* ap  = A  + (long)(tm * 16 + llo) * fi + lhi * 16;
    // B fragments: lane col = (tb*64 + j*16 + llo); elems 0-15 -> K = k + lhi*16 + e
    const unsigned short* bp0 = Bt + (long)(tb * 64 + llo) * fi + lhi * 16;
    const unsigned short* bp1 = bp0 + (long)16 * fi;
    const unsigned short* bp2 = bp0 + (long)32 * fi;
    const unsigned short* bp3 = bp0 + (long)48 * fi;

    v8f acc0 = {}, acc1 = {}, acc2 = {}, acc3 = {};
    for (int k = 0; k < fi; k += 32) {
        union { v16bf v; u32x4 q[2]; } a, b0, b1, b2, b3;
        a.q[0]  = *(const u32x4*)(ap + k);
        a.q[1]  = *(const u32x4*)(ap + k + 8);
        b0.q[0] = *(const u32x4*)(bp0 + k);
        b0.q[1] = *(const u32x4*)(bp0 + k + 8);
        b1.q[0] = *(const u32x4*)(bp1 + k);
        b1.q[1] = *(const u32x4*)(bp1 + k + 8);
        b2.q[0] = *(const u32x4*)(bp2 + k);
        b2.q[1] = *(const u32x4*)(bp2 + k + 8);
        b3.q[0] = *(const u32x4*)(bp3 + k);
        b3.q[1] = *(const u32x4*)(bp3 + k + 8);
        acc0 = __builtin_amdgcn_wmma_f32_16x16x32_bf16(false, a.v, false, b0.v, (short)0, acc0, false, false);
        acc1 = __builtin_amdgcn_wmma_f32_16x16x32_bf16(false, a.v, false, b1.v, (short)0, acc1, false, false);
        acc2 = __builtin_amdgcn_wmma_f32_16x16x32_bf16(false, a.v, false, b2.v, (short)0, acc2, false, false);
        acc3 = __builtin_amdgcn_wmma_f32_16x16x32_bf16(false, a.v, false, b3.v, (short)0, acc3, false, false);
    }

    // C/D layout: VGPR v -> row = tm*16 + lhi*8 + v, col = block col + llo
    float* hp = H + (long)(tm * 16 + lhi * 8) * fo + tb * 64 + llo;
#pragma unroll
    for (int v = 0; v < 8; ++v) {
        hp[(long)v * fo +  0] = acc0[v];
        hp[(long)v * fo + 16] = acc1[v];
        hp[(long)v * fo + 32] = acc2[v];
        hp[(long)v * fo + 48] = acc3[v];
    }
}

// OUT init fused with self-loop: OUT[i,f] = H[i,f] * dinv[i]^2
__global__ void k_init_out(const float* __restrict__ H, const float* __restrict__ dinv,
                           float* __restrict__ OUT, int n, int fo) {
    long t = blockIdx.x * (long)blockDim.x + threadIdx.x;
    long total = (long)n * fo;
    if (t >= total) return;
    int i = (int)(t / fo);
    float di = dinv[i];
    OUT[t] = H[t] * di * di;
}

// scatter-add: OUT[dst] += H[src] * dinv[src]*dinv[dst], thread per (edge, 4 feats)
__global__ void k_scatter(const float* __restrict__ H, const int* __restrict__ src,
                          const int* __restrict__ dst, const float* __restrict__ dinv,
                          float* __restrict__ OUT, int E, int fo) {
    long t = blockIdx.x * (long)blockDim.x + threadIdx.x;
    int  pe = fo >> 2;
    long total = (long)E * pe;
    if (t >= total) return;
    int e = (int)(t / pe);
    int f = (int)(t - (long)e * pe) << 2;
    int s = src[e], d = dst[e];
    float w = dinv[s] * dinv[d];
    const float* hp = H   + (long)s * fo + f;
    float*       op = OUT + (long)d * fo + f;
    atomicAdd(op + 0, hp[0] * w);
    atomicAdd(op + 1, hp[1] * w);
    atomicAdd(op + 2, hp[2] * w);
    atomicAdd(op + 3, hp[3] * w);
}

// bias + ReLU; emit bf16 next-layer input and/or fp32 final output
__global__ void k_finalize(const float* __restrict__ OUT, const float* __restrict__ bias,
                           unsigned short* __restrict__ xb_next, float* __restrict__ out_f32,
                           int n, int fo, int do_relu) {
    long t = blockIdx.x * (long)blockDim.x + threadIdx.x;
    long total = (long)n * fo;
    if (t >= total) return;
    int f = (int)(t % fo);
    float v = OUT[t] + bias[f];
    if (do_relu) v = fmaxf(v, 0.0f);
    if (xb_next) xb_next[t] = f2bf(v);
    if (out_f32) out_f32[t] = v;
}

extern "C" void kernel_launch(void* const* d_in, const int* in_sizes, int n_in,
                              void* d_out, int out_size, void* d_ws, size_t ws_size,
                              hipStream_t stream) {
    (void)n_in; (void)out_size; (void)ws_size;
    const int DIMS[6] = {128, 256, 256, 256, 256, 64};

    const float* x  = (const float*)d_in[0];
    const int*   ei = (const int*)d_in[1];
    int N = in_sizes[0] / DIMS[0];
    int E = in_sizes[1] / 2;
    const int* src = ei;
    const int* dst = ei + E;

    char* ws = (char*)d_ws;
    size_t off = 0;
    auto alloc = [&](size_t bytes) -> void* {
        void* p = ws + off;
        off += (bytes + 255) & ~(size_t)255;
        return p;
    };
    float*          deg  = (float*)         alloc((size_t)N * 4);
    float*          dinv = (float*)         alloc((size_t)N * 4);
    unsigned short* Xb   = (unsigned short*)alloc((size_t)N * 256 * 2);
    float*          H    = (float*)         alloc((size_t)N * 256 * 4);
    float*          OUT  = (float*)         alloc((size_t)N * 256 * 4);
    unsigned short* Wt   = (unsigned short*)alloc((size_t)256 * 256 * 2);

    const int BT = 256;

    // normalization: deg (in-degree + self-loop) -> dinv = rsqrt(deg)
    k_zero  <<<dim3((N + BT - 1) / BT), dim3(BT), 0, stream>>>(deg, (long)N);
    k_degree<<<dim3((E + BT - 1) / BT), dim3(BT), 0, stream>>>(dst, deg, E);
    k_dinv  <<<dim3((N + BT - 1) / BT), dim3(BT), 0, stream>>>(deg, dinv, N);

    // x -> bf16
    long nx = (long)N * DIMS[0];
    k_cvt_bf16<<<dim3((unsigned)((nx + BT - 1) / BT)), dim3(BT), 0, stream>>>(x, Xb, nx);

    for (int l = 0; l < 5; ++l) {
        int fi = DIMS[l], fo = DIMS[l + 1];
        const float* W = (const float*)d_in[2 + 2 * l];
        const float* b = (const float*)d_in[3 + 2 * l];

        int wcnt = fi * fo;
        k_cvt_w<<<dim3((wcnt + BT - 1) / BT), dim3(BT), 0, stream>>>(W, Wt, fi, fo);

        long waves = (long)(N / 16) * (fo / 64);   // one wave per 16x64 tile
        long thr   = waves * 32;
        k_gemm_wmma<<<dim3((unsigned)((thr + BT - 1) / BT)), dim3(BT), 0, stream>>>(
            Xb, Wt, H, N, fi, fo);

        long no = (long)N * fo;
        k_init_out<<<dim3((unsigned)((no + BT - 1) / BT)), dim3(BT), 0, stream>>>(
            H, dinv, OUT, N, fo);

        long st = (long)E * (fo >> 2);
        k_scatter<<<dim3((unsigned)((st + BT - 1) / BT)), dim3(BT), 0, stream>>>(
            H, src, dst, dinv, OUT, E, fo);

        int             relu = (l < 4) ? 1 : 0;
        unsigned short* nxb  = (l < 4) ? Xb : nullptr;
        float*          of   = (l == 4) ? (float*)d_out : nullptr;
        k_finalize<<<dim3((unsigned)((no + BT - 1) / BT)), dim3(BT), 0, stream>>>(
            OUT, b, nxb, of, N, fo, relu);
    }
}
